// StableCrossAttention_27865747816825
// MI455X (gfx1250) — compile-verified, matrix operand order checked by reading, not run on hardware
//
#include <hip/hip_runtime.h>

#define D_FEAT 1024
#define N_HEADS 16
#define HEAD_DIM 64
#define SEQ 2048
#define BATCH 2
#define MTOK (BATCH * SEQ) // 4096

typedef __attribute__((ext_vector_type(16))) _Float16 v16h;
typedef __attribute__((ext_vector_type(8)))  _Float16 v8h;
typedef __attribute__((ext_vector_type(8)))  float    v8f;

// b128 transfer unit for the async-to-LDS builtin (type from probe diagnostic)
typedef int i128v __attribute__((vector_size(16)));
typedef __attribute__((address_space(1))) i128v* gbl_b128_p;
typedef __attribute__((address_space(3))) i128v* lds_b128_p;

#if __has_builtin(__builtin_amdgcn_global_load_async_to_lds_b128) && \
    __has_builtin(__builtin_amdgcn_s_wait_asynccnt)
#define USE_ASYNC_LDS 1
#else
#define USE_ASYNC_LDS 0
#endif

static __device__ __forceinline__ v8f wmma_f16(v16h a, v16h b, v8f c) {
  // D = A(16x32 f16) * B(32x16 f16) + C(16x16 f32)
  return __builtin_amdgcn_wmma_f32_16x16x32_f16(
      /*neg_a=*/false, a, /*neg_b=*/false, b,
      /*c_mod=*/(short)0, c, /*reuse_a=*/false, /*reuse_b=*/false);
}

// Load one 16x32 f16 operand fragment from a row-major tile with leading dim
// `ld` (elements). Per the CDNA5 VGPR layout, lane&15 selects the row and
// lanes 16..31 hold the K={8..15,24..31} half -> two contiguous 16B loads.
static __device__ __forceinline__ v16h load_frag(const _Float16* __restrict__ tile, int ld) {
  const int lane = threadIdx.x & 31;
  const int r    = lane & 15;
  const int k0   = (lane & 16) ? 8 : 0;
  const _Float16* p = tile + (size_t)r * ld + k0;
  v8h lo = *(const v8h*)(p);
  v8h hi = *(const v8h*)(p + 16);
  return __builtin_shufflevector(lo, hi, 0,1,2,3,4,5,6,7,8,9,10,11,12,13,14,15);
}

// Move 16 f16 (32B) global -> LDS: async DMA path if available, else direct.
static __device__ __forceinline__ void stage16(const _Float16* g, _Float16* l) {
#if USE_ASYNC_LDS
  __builtin_amdgcn_global_load_async_to_lds_b128((gbl_b128_p)g, (lds_b128_p)l, 0, 0);
  __builtin_amdgcn_global_load_async_to_lds_b128((gbl_b128_p)(g + 8), (lds_b128_p)(l + 8), 0, 0);
#else
  *(v8h*)l = *(const v8h*)g;
  *(v8h*)(l + 8) = *(const v8h*)(g + 8);
#endif
}

// ---------------------------------------------------------------------------
// f32 -> f16 elementwise convert (for weights)
// ---------------------------------------------------------------------------
__global__ __launch_bounds__(256) void cvt_f32_to_f16(const float* __restrict__ src,
                                                      _Float16* __restrict__ dst, int n) {
  int i = blockIdx.x * blockDim.x + threadIdx.x;
  if (i < n) dst[i] = (_Float16)src[i];
}

// ---------------------------------------------------------------------------
// LayerNorm over D=1024, output f16. One row per 256-thread block (8 waves).
// ---------------------------------------------------------------------------
__global__ __launch_bounds__(256) void ln_f16_kernel(const float* __restrict__ x,
                                                     const float* __restrict__ gamma,
                                                     const float* __restrict__ beta,
                                                     _Float16* __restrict__ xn) {
  const int row  = blockIdx.x;
  const int tid  = threadIdx.x;
  const int wave = tid >> 5, lane = tid & 31;
  const float* xr = x + (size_t)row * D_FEAT;
  float4 v = *(const float4*)(xr + tid * 4);

  float s = v.x + v.y + v.z + v.w;
  #pragma unroll
  for (int off = 16; off; off >>= 1) s += __shfl_xor(s, off, 32);
  __shared__ float red0[8], red1[8];
  if (lane == 0) red0[wave] = s;
  __syncthreads();
  float tot = 0.f;
  #pragma unroll
  for (int i = 0; i < 8; ++i) tot += red0[i];
  const float mu = tot * (1.0f / D_FEAT);

  float dx = v.x - mu, dy = v.y - mu, dz = v.z - mu, dw = v.w - mu;
  float s2 = dx * dx + dy * dy + dz * dz + dw * dw;
  #pragma unroll
  for (int off = 16; off; off >>= 1) s2 += __shfl_xor(s2, off, 32);
  if (lane == 0) red1[wave] = s2;
  __syncthreads();
  float tot2 = 0.f;
  #pragma unroll
  for (int i = 0; i < 8; ++i) tot2 += red1[i];
  const float rstd = rsqrtf(tot2 * (1.0f / D_FEAT) + 1e-5f);

  const int c = tid * 4;
  _Float16* o = xn + (size_t)row * D_FEAT + c;
  o[0] = (_Float16)(dx * rstd * gamma[c + 0] + beta[c + 0]);
  o[1] = (_Float16)(dy * rstd * gamma[c + 1] + beta[c + 1]);
  o[2] = (_Float16)(dz * rstd * gamma[c + 2] + beta[c + 2]);
  o[3] = (_Float16)(dw * rstd * gamma[c + 3] + beta[c + 3]);
}

// ---------------------------------------------------------------------------
// WMMA GEMM: C[M,N] = A[M,K] * W[N,K]^T  (f16 in, row-major, ld=K).
// Block = 128 threads = 4 waves; block tile 64x128; wave tile 32x64
// (2x4 fragments -> 8 WMMAs per 6 fragment loads). K-loop manually unrolled
// 2x with ping-pong fragment registers: zero rotation movs, next-tile loads
// always in flight behind the current tile's 8 WMMAs.
// mode 0: scatter f16 -> [B,H,M,Hd]   (Q, K)
// mode 1: scatter f16 -> [B,H,Hd,M]   (V transposed, attention B-operand)
// mode 2: f32 out = resid + 0.1*(acc + bias)   (final projection)
// ---------------------------------------------------------------------------
__global__ __launch_bounds__(128) void gemm_xwt_kernel(const _Float16* __restrict__ A,
                                                       const _Float16* __restrict__ W,
                                                       const float* __restrict__ resid,
                                                       const float* __restrict__ bias,
                                                       _Float16* __restrict__ out16,
                                                       float* __restrict__ out32,
                                                       int M, int N, int K, int mode) {
  const int wave = threadIdx.x >> 5;
  const int lane = threadIdx.x & 31;
  const int m0 = blockIdx.x * 64 + (wave >> 1) * 32;
  const int n0 = blockIdx.y * 128 + (wave & 1) * 64;

  const _Float16* A0 = A + (size_t)m0 * K;
  const _Float16* A1 = A + (size_t)(m0 + 16) * K;
  const _Float16* W0 = W + (size_t)(n0 +  0) * K;
  const _Float16* W1 = W + (size_t)(n0 + 16) * K;
  const _Float16* W2 = W + (size_t)(n0 + 32) * K;
  const _Float16* W3 = W + (size_t)(n0 + 48) * K;

  v8f acc[2][4] = {};

  // prologue: fragments for k = 0 into the "c" bank
  v16h ca0 = load_frag(A0, K), ca1 = load_frag(A1, K);
  v16h cb0 = load_frag(W0, K), cb1 = load_frag(W1, K);
  v16h cb2 = load_frag(W2, K), cb3 = load_frag(W3, K);

  #pragma unroll 1
  for (int k = 0; k < K; k += 64) {   // K % 64 == 0
    const int k1 = k + 32;            // always < K
    if (k + 128 <= K) {
      __builtin_prefetch(A0 + k + 64, 0, 1);
      __builtin_prefetch(W0 + k + 64, 0, 1);
      __builtin_prefetch(W2 + k + 64, 0, 1);
    }
    // load k+32 into "n" bank, compute k from "c" bank
    v16h na0 = load_frag(A0 + k1, K), na1 = load_frag(A1 + k1, K);
    v16h nb0 = load_frag(W0 + k1, K), nb1 = load_frag(W1 + k1, K);
    v16h nb2 = load_frag(W2 + k1, K), nb3 = load_frag(W3 + k1, K);

    acc[0][0] = wmma_f16(ca0, cb0, acc[0][0]);
    acc[0][1] = wmma_f16(ca0, cb1, acc[0][1]);
    acc[0][2] = wmma_f16(ca0, cb2, acc[0][2]);
    acc[0][3] = wmma_f16(ca0, cb3, acc[0][3]);
    acc[1][0] = wmma_f16(ca1, cb0, acc[1][0]);
    acc[1][1] = wmma_f16(ca1, cb1, acc[1][1]);
    acc[1][2] = wmma_f16(ca1, cb2, acc[1][2]);
    acc[1][3] = wmma_f16(ca1, cb3, acc[1][3]);

    // load k+64 into "c" bank (wrapped dummy on final trip), compute k+32
    const int k2 = (k + 64 < K) ? (k + 64) : 0;
    ca0 = load_frag(A0 + k2, K); ca1 = load_frag(A1 + k2, K);
    cb0 = load_frag(W0 + k2, K); cb1 = load_frag(W1 + k2, K);
    cb2 = load_frag(W2 + k2, K); cb3 = load_frag(W3 + k2, K);

    acc[0][0] = wmma_f16(na0, nb0, acc[0][0]);
    acc[0][1] = wmma_f16(na0, nb1, acc[0][1]);
    acc[0][2] = wmma_f16(na0, nb2, acc[0][2]);
    acc[0][3] = wmma_f16(na0, nb3, acc[0][3]);
    acc[1][0] = wmma_f16(na1, nb0, acc[1][0]);
    acc[1][1] = wmma_f16(na1, nb1, acc[1][1]);
    acc[1][2] = wmma_f16(na1, nb2, acc[1][2]);
    acc[1][3] = wmma_f16(na1, nb3, acc[1][3]);
  }

  // C fragment layout: element (m = j + 8*(lane>=16), n = lane&15) in VGPR j.
  const int gq = (lane & 16) ? 8 : 0;
  const int nc = lane & 15;
  #pragma unroll
  for (int mi = 0; mi < 2; ++mi) {
    #pragma unroll
    for (int ni = 0; ni < 4; ++ni) {
      #pragma unroll
      for (int j = 0; j < 8; ++j) {
        const int row = m0 + mi * 16 + j + gq;   // token index
        const int col = n0 + ni * 16 + nc;       // feature index
        const float v = acc[mi][ni][j];
        if (mode == 2) {
          out32[(size_t)row * D_FEAT + col] =
              resid[(size_t)row * D_FEAT + col] + 0.1f * (v + bias[col]);
        } else {
          const int b = row >> 11, m = row & (SEQ - 1);
          const int h = col >> 6,  d = col & (HEAD_DIM - 1);
          size_t idx;
          if (mode == 0)    idx = (((size_t)(b * N_HEADS + h) * SEQ + m) * HEAD_DIM) + d;
          else /* mode 1 */ idx = (((size_t)(b * N_HEADS + h) * HEAD_DIM + d) * SEQ) + m;
          out16[idx] = (_Float16)v;
        }
      }
    }
  }
}

// ---------------------------------------------------------------------------
// Flash-style attention. Q,K: [B*H, SEQ, 64] f16; Vt: [B*H, 64, SEQ] f16.
// Block = 128 threads = 4 waves; each wave owns 16 query rows (block: 64).
// K/V tiles for each 32-key step are staged cooperatively into double-
// buffered LDS (async-to-LDS when available) -- removes the 4x per-wave
// redundant global fragment loads. Softmax P transposes through per-wave LDS.
// scale = Hd^-0.5 * 0.5 = 0.0625, scores clipped to [-10,10].
// ---------------------------------------------------------------------------
__global__ __launch_bounds__(128) void attn_kernel(const _Float16* __restrict__ Q,
                                                   const _Float16* __restrict__ Kh,
                                                   const _Float16* __restrict__ Vt,
                                                   _Float16* __restrict__ Oout) {
  const int qt   = blockIdx.x & (SEQ / 64 - 1);  // query tile within head
  const int bh   = blockIdx.x / (SEQ / 64);      // fused batch*head
  const int tid  = threadIdx.x;
  const int wave = tid >> 5;
  const int lane = tid & 31;
  const int qrow = qt * 64 + wave * 16;

  const _Float16* Qb = Q  + ((size_t)bh * SEQ + qrow) * HEAD_DIM;
  const _Float16* Kb = Kh + (size_t)bh * SEQ * HEAD_DIM;
  const _Float16* Vb = Vt + (size_t)bh * HEAD_DIM * SEQ;

  __shared__ __attribute__((aligned(16))) _Float16 lds_k[2][32 * HEAD_DIM]; // 2x4KB
  __shared__ __attribute__((aligned(16))) _Float16 lds_v[2][HEAD_DIM * 32]; // 2x4KB
  __shared__ __attribute__((aligned(16))) _Float16 lds_p[4][16 * 32];       // 4x1KB
  _Float16* Pt = lds_p[wave];

  // staging map: K tile is one contiguous 2048-half slab (row stride 64);
  // V tile rows (head dims) are strided by SEQ, 32 halves per row.
  const int vrow = tid >> 1, vcol = (tid & 1) * 16;

  const v16h qa0 = load_frag(Qb, HEAD_DIM);       // head dims 0..31
  const v16h qa1 = load_frag(Qb + 32, HEAD_DIM);  // head dims 32..63

  v8f o0 = {}, o1 = {}, o2 = {}, o3 = {};
  float mrow[8], lrow[8];
  #pragma unroll
  for (int j = 0; j < 8; ++j) { mrow[j] = -1e30f; lrow[j] = 0.f; }

  const int gq = (lane & 16) ? 8 : 0;
  const int nc = lane & 15;
  const float scale = 0.0625f;

  // stage tile at key offset kt into buffer b (4 async b128 ops per thread)
  auto stage_tiles = [&](int b, int kt) {
    stage16(Kb + (size_t)kt * HEAD_DIM + tid * 16, &lds_k[b][tid * 16]);
    stage16(Vb + (size_t)vrow * SEQ + kt + vcol, &lds_v[b][vrow * 32 + vcol]);
  };

  stage_tiles(0, 0);

  for (int kt = 0; kt < SEQ; kt += 32) {
    const int buf = (kt >> 5) & 1;
    if (kt + 32 < SEQ) {
      stage_tiles(buf ^ 1, kt + 32);
#if USE_ASYNC_LDS
      __builtin_amdgcn_s_wait_asynccnt(4);  // current tile's 4 ops retired
#endif
    } else {
#if USE_ASYNC_LDS
      __builtin_amdgcn_s_wait_asynccnt(0);
#endif
    }
    __syncthreads();  // K/V tile visible block-wide

    // ---- scores: S[16q x 32k] as two 16x16 f32 fragments (from LDS) -------
    v8f s0 = {}, s1 = {};
    {
      v16h kb = load_frag(&lds_k[buf][0], HEAD_DIM);
      s0 = wmma_f16(qa0, kb, s0);
      kb = load_frag(&lds_k[buf][32], HEAD_DIM);
      s0 = wmma_f16(qa1, kb, s0);
      kb = load_frag(&lds_k[buf][16 * HEAD_DIM], HEAD_DIM);
      s1 = wmma_f16(qa0, kb, s1);
      kb = load_frag(&lds_k[buf][16 * HEAD_DIM + 32], HEAD_DIM);
      s1 = wmma_f16(qa1, kb, s1);
    }
    // scale + clip + tile row-max
    float tmax[8];
    #pragma unroll
    for (int j = 0; j < 8; ++j) {
      float a = fminf(fmaxf(s0[j] * scale, -10.f), 10.f); s0[j] = a;
      float b = fminf(fmaxf(s1[j] * scale, -10.f), 10.f); s1[j] = b;
      tmax[j] = fmaxf(a, b);
    }
    #pragma unroll
    for (int off = 1; off < 16; off <<= 1) {
      #pragma unroll
      for (int j = 0; j < 8; ++j)
        tmax[j] = fmaxf(tmax[j], __shfl_xor(tmax[j], off, 32));
    }
    // online-softmax update
    float corr[8], psum[8];
    #pragma unroll
    for (int j = 0; j < 8; ++j) {
      const float mnew = fmaxf(mrow[j], tmax[j]);
      corr[j] = __expf(mrow[j] - mnew);
      mrow[j] = mnew;
      const float p0 = __expf(s0[j] - mnew);
      const float p1 = __expf(s1[j] - mnew);
      psum[j] = p0 + p1;
      Pt[(j + gq) * 32 + nc]      = (_Float16)p0;  // P row-major 16x32 in LDS
      Pt[(j + gq) * 32 + nc + 16] = (_Float16)p1;
    }
    #pragma unroll
    for (int off = 1; off < 16; off <<= 1) {
      #pragma unroll
      for (int j = 0; j < 8; ++j) psum[j] += __shfl_xor(psum[j], off, 32);
    }
    #pragma unroll
    for (int j = 0; j < 8; ++j) {
      lrow[j] = lrow[j] * corr[j] + psum[j];
      o0[j] *= corr[j]; o1[j] *= corr[j]; o2[j] *= corr[j]; o3[j] *= corr[j];
    }
    __syncthreads();  // publish P (C-layout -> A-layout via LDS)
    const v16h pa = load_frag(Pt, 32);
    {
      v16h vb = load_frag(&lds_v[buf][0 * 16 * 32], 32);
      o0 = wmma_f16(pa, vb, o0);
      vb = load_frag(&lds_v[buf][1 * 16 * 32], 32);
      o1 = wmma_f16(pa, vb, o1);
      vb = load_frag(&lds_v[buf][2 * 16 * 32], 32);
      o2 = wmma_f16(pa, vb, o2);
      vb = load_frag(&lds_v[buf][3 * 16 * 32], 32);
      o3 = wmma_f16(pa, vb, o3);
    }
    __syncthreads();  // all reads of buf & P done before next overwrite
  }

  // normalize + scatter into [token, D_FEAT] f16
  const int b = bh >> 4, h = bh & 15;
  #pragma unroll
  for (int j = 0; j < 8; ++j) {
    const float inv = 1.0f / lrow[j];
    const size_t token = (size_t)b * SEQ + qrow + j + gq;
    _Float16* orow = Oout + token * D_FEAT + h * HEAD_DIM + nc;
    orow[0]  = (_Float16)(o0[j] * inv);
    orow[16] = (_Float16)(o1[j] * inv);
    orow[32] = (_Float16)(o2[j] * inv);
    orow[48] = (_Float16)(o3[j] * inv);
  }
}

// ---------------------------------------------------------------------------
extern "C" void kernel_launch(void* const* d_in, const int* in_sizes, int n_in,
                              void* d_out, int out_size, void* d_ws, size_t ws_size,
                              hipStream_t stream) {
  const float* x  = (const float*)d_in[0];
  const float* Wq = (const float*)d_in[1];
  const float* Wk = (const float*)d_in[2];
  const float* Wv = (const float*)d_in[3];
  const float* Wo = (const float*)d_in[4];
  const float* bo = (const float*)d_in[5];
  const float* g  = (const float*)d_in[6];
  const float* be = (const float*)d_in[7];
  float* out = (float*)d_out;

  char* ws = (char*)d_ws;
  const size_t MB = 1u << 20;
  _Float16* xn   = (_Float16*)(ws + 0);        // [4096,1024] f16   (8 MiB)
  _Float16* wq16 = (_Float16*)(ws + 8  * MB);  // [1024,1024] f16   (2 MiB each)
  _Float16* wk16 = (_Float16*)(ws + 10 * MB);
  _Float16* wv16 = (_Float16*)(ws + 12 * MB);
  _Float16* wo16 = (_Float16*)(ws + 14 * MB);
  _Float16* q16  = (_Float16*)(ws + 16 * MB);  // [B,H,SEQ,64]      (8 MiB)
  _Float16* k16  = (_Float16*)(ws + 24 * MB);  // [B,H,SEQ,64]      (8 MiB)
  _Float16* vt16 = (_Float16*)(ws + 32 * MB);  // [B,H,64,SEQ]      (8 MiB)
  _Float16* ao16 = (_Float16*)(ws + 40 * MB);  // [4096,1024]       (8 MiB)

  const int wn = D_FEAT * D_FEAT;
  cvt_f32_to_f16<<<(wn + 255) / 256, 256, 0, stream>>>(Wq, wq16, wn);
  cvt_f32_to_f16<<<(wn + 255) / 256, 256, 0, stream>>>(Wk, wk16, wn);
  cvt_f32_to_f16<<<(wn + 255) / 256, 256, 0, stream>>>(Wv, wv16, wn);
  cvt_f32_to_f16<<<(wn + 255) / 256, 256, 0, stream>>>(Wo, wo16, wn);

  ln_f16_kernel<<<MTOK, 256, 0, stream>>>(x, g, be, xn);

  dim3 ggrid(MTOK / 64, D_FEAT / 128);
  gemm_xwt_kernel<<<ggrid, 128, 0, stream>>>(xn, wq16, nullptr, nullptr, q16, nullptr,
                                             MTOK, D_FEAT, D_FEAT, 0);
  gemm_xwt_kernel<<<ggrid, 128, 0, stream>>>(xn, wk16, nullptr, nullptr, k16, nullptr,
                                             MTOK, D_FEAT, D_FEAT, 0);
  gemm_xwt_kernel<<<ggrid, 128, 0, stream>>>(xn, wv16, nullptr, nullptr, vt16, nullptr,
                                             MTOK, D_FEAT, D_FEAT, 1);

  attn_kernel<<<BATCH * N_HEADS * (SEQ / 64), 128, 0, stream>>>(q16, k16, vt16, ao16);

  gemm_xwt_kernel<<<ggrid, 128, 0, stream>>>(ao16, wo16, x, bo, nullptr, out,
                                             MTOK, D_FEAT, D_FEAT, 2);
}
